// Attention_66829691126159
// MI455X (gfx1250) — matrix operand order repaired and checked by reading, not compiled, not run on hardware
//
#include <hip/hip_runtime.h>
#include <hip/hip_bf16.h>
#include <stdint.h>

// ---------------------------------------------------------------------------
// GQA attention (B=2,T=2048,C=2048,HQ=16,KVH=8,D=128) for gfx1250 (MI455X).
// All GEMMs + attention run on v_wmma_f32_16x16x32_bf16 (fp32 accumulate).
// Flash-style attention: the [T,T] logits never touch HBM.
// ---------------------------------------------------------------------------

typedef __attribute__((ext_vector_type(16))) __bf16 v16bf;
typedef __attribute__((ext_vector_type(8)))  float  v8f;
typedef __attribute__((ext_vector_type(4)))  float  v4f;
typedef __attribute__((ext_vector_type(4)))  __bf16 v4bf;
typedef __attribute__((ext_vector_type(4)))  unsigned int u32x4;

union Frag16 { v16bf v; u32x4 q[2]; };
union Pack8  { u32x4 q; __bf16 e[8]; };

#define WMMA_BF16(A_, B_, C_) \
  __builtin_amdgcn_wmma_f32_16x16x32_bf16(false, (A_), false, (B_), (short)0, (C_), false, false)

// ------------------------------ fp32 -> bf16 -------------------------------
__global__ __launch_bounds__(256) void cvt_f32_to_bf16(const float* __restrict__ src,
                                                       __bf16* __restrict__ dst, int n4) {
  int i = blockIdx.x * blockDim.x + threadIdx.x;
  if (i < n4) {
    v4f v = ((const v4f*)src)[i];
    ((v4bf*)dst)[i] = __builtin_convertvector(v, v4bf);
  }
}

// ------------------------------ RoPE (in place) ----------------------------
// layout: P[rows][nheads*128], pair (d, d+64) rotated by angle t / 1e6^(d/64)
__global__ __launch_bounds__(256) void rope_inplace(__bf16* __restrict__ P,
                                                    int nheads, int rows, int T) {
  int idx = blockIdx.x * blockDim.x + threadIdx.x;
  int total = rows * nheads * 64;
  if (idx >= total) return;
  int d   = idx & 63;
  int tmp = idx >> 6;
  int h   = tmp % nheads;
  int row = tmp / nheads;
  int t   = row % T;
  // 1e6^(-d/64) = exp(-d * ln(1e6)/64)
  float inv_ts = __expf(-(float)d * (13.815510557964274f / 64.0f));
  float ang = (float)t * inv_ts;
  float s, c;
  __sincosf(ang, &s, &c);
  size_t base = (size_t)row * (size_t)(nheads * 128) + (size_t)h * 128 + d;
  float u1 = (float)P[base];
  float u2 = (float)P[base + 64];
  P[base]      = (__bf16)(u1 * c - u2 * s);
  P[base + 64] = (__bf16)(u2 * c + u1 * s);
}

// ------------------------------ WMMA GEMM ----------------------------------
// C[M,N] = A[M,Kd] * Bw[Kd,N], A/Bw bf16 row-major.
// Block: 256 thr = 8 waves, tile 128(M) x 128(N); wave tile 32 x 64; K step 32.
// 8 WMMAs per wave per K-step (two independent per B-fragment -> LDS latency
// hiding), 64 accumulator VGPRs per lane.
template <bool OUT_BF16>
__global__ __launch_bounds__(256) void gemm_bf16_wmma(const __bf16* __restrict__ A,
                                                      const __bf16* __restrict__ Bw,
                                                      void* __restrict__ Cp,
                                                      int N, int Kd) {
  __shared__ __bf16 Wt[128][40];   // transposed B tile: [n_local][k], stride 40 keeps 16B align

  const int tid  = threadIdx.x;
  const int lane = tid & 31;
  const int wave = tid >> 5;
  const int l16  = lane & 15;
  const int hi   = lane >> 4;
  const int wm   = wave & 3;               // 0..3  -> 4 x 32 rows
  const int wn   = wave >> 2;              // 0..1  -> 2 x 64 cols
  const int m0   = blockIdx.x * 128 + wm * 32;
  const int n0b  = blockIdx.y * 128;
  const int n0w  = wn * 64;
  const int aoff = hi * 8;                 // A-frag K sub-offset per lane group
  const int koff = hi * 16;                // unused; B-frag uses aoff, same k order as A

  const __bf16* Arow0 = A + (size_t)(m0 + l16) * (size_t)Kd;
  const __bf16* Arow1 = Arow0 + (size_t)16 * (size_t)Kd;
  const int srow = tid >> 3;               // 0..31  (k row of staged tile)
  const int scol = (tid & 7) * 16;         // 0..112 (n col of staged tile)

  v8f acc0[4] = {};
  v8f acc1[4] = {};

  for (int kb = 0; kb < Kd; kb += 32) {
    __syncthreads();
    // stage 32x128 B tile transposed into LDS
    const __bf16* bp = Bw + (size_t)(kb + srow) * (size_t)N + n0b + scol;
    Pack8 p0, p1;
    p0.q = *(const u32x4*)bp;
    p1.q = *(const u32x4*)(bp + 8);
    __builtin_prefetch(bp + (size_t)32 * N, 0, 1);
#pragma unroll
    for (int j = 0; j < 8; j++) {
      Wt[scol + j][srow]     = p0.e[j];
      Wt[scol + 8 + j][srow] = p1.e[j];
    }
    __syncthreads();

    Frag16 a0, a1;
    a0.q[0] = *(const u32x4*)(Arow0 + kb + aoff);
    a0.q[1] = *(const u32x4*)(Arow0 + kb + 16 + aoff);
    a1.q[0] = *(const u32x4*)(Arow1 + kb + aoff);
    a1.q[1] = *(const u32x4*)(Arow1 + kb + 16 + aoff);

    Frag16 bfr[4];
#pragma unroll
    for (int nt = 0; nt < 4; nt++) {
      const __bf16* wp = &Wt[n0w + nt * 16 + l16][aoff];
      bfr[nt].q[0] = *(const u32x4*)wp;
      bfr[nt].q[1] = *(const u32x4*)(wp + 16);
    }
#pragma unroll
    for (int nt = 0; nt < 4; nt++) {
      acc0[nt] = WMMA_BF16(a0.v, bfr[nt].v, acc0[nt]);
      acc1[nt] = WMMA_BF16(a1.v, bfr[nt].v, acc1[nt]);
    }
  }

  // epilogue: straight-line stores (no runtime branch; template-resolved type)
#pragma unroll
  for (int nt = 0; nt < 4; nt++) {
#pragma unroll
    for (int r = 0; r < 8; r++) {
      int row0 = m0 + r + 8 * hi;
      int col  = n0b + n0w + nt * 16 + l16;
      if (OUT_BF16) {
        __bf16* Cb = (__bf16*)Cp;
        Cb[(size_t)row0 * N + col]        = (__bf16)acc0[nt][r];
        Cb[(size_t)(row0 + 16) * N + col] = (__bf16)acc1[nt][r];
      } else {
        float* Cf = (float*)Cp;
        Cf[(size_t)row0 * N + col]        = acc0[nt][r];
        Cf[(size_t)(row0 + 16) * N + col] = acc1[nt][r];
      }
    }
  }
}

// --------------------------- Flash attention -------------------------------
// grid: (T/64, B*HQ). block 128 thr = 4 waves, each wave owns a 16-query tile.
// Per key block of 32: S = Q K^T (8 WMMA), softcap+mask, online softmax,
// rescale O, O += P V (8 WMMA with V^T staged in LDS).
__global__ __launch_bounds__(128) void flash_attn_wmma(const __bf16* __restrict__ Q,
                                                       const __bf16* __restrict__ Kg,
                                                       const __bf16* __restrict__ Vg,
                                                       __bf16* __restrict__ O) {
  const int T = 2048;
  __shared__ float  Sbuf[4][16][32];
  __shared__ __bf16 Pbuf[4][16][32];
  __shared__ __bf16 VtS[4][128][40];       // V^T tile per wave: [d][k], stride 40
  __shared__ float  Stats[4][2][16];       // [0]=alpha, [1]=1/l

  const int tid  = threadIdx.x;
  const int lane = tid & 31;
  const int wave = tid >> 5;
  const int l16  = lane & 15;
  const int hi   = lane >> 4;
  const int head = blockIdx.y;             // b*16 + qh
  const int b    = head >> 4;
  const int qh   = head & 15;
  const int kvh  = qh & 7;                 // qh = r*8 + h  ->  kv head = qh % 8
  const int s0   = blockIdx.x * 64 + wave * 16;
  const int aoff = hi * 8;
  const int koff = hi * 16;

  const __bf16* Qrow = Q  + (size_t)(b * T + s0 + l16) * 2048 + qh * 128;
  const __bf16* Kb   = Kg + (size_t)(b * T) * 1024 + kvh * 128;
  const __bf16* Vb   = Vg + (size_t)(b * T) * 1024 + kvh * 128;

  // Q A-fragments for the whole head_dim (4 chunks of K=32)
  Frag16 aq[4];
#pragma unroll
  for (int kc = 0; kc < 4; kc++) {
    aq[kc].q[0] = *(const u32x4*)(Qrow + kc * 32 + aoff);
    aq[kc].q[1] = *(const u32x4*)(Qrow + kc * 32 + 16 + aoff);
  }

  v8f o[8] = {};
  float m_r = -3.0e38f, l_r = 0.0f;
  const float scale = 0.08838834764831845f;    // 1/sqrt(128)
  const int kend = blockIdx.x * 64 + 63;       // uniform per block -> barriers legal

  for (int kb = 0; kb <= kend; kb += 32) {
    __syncthreads();
    // ---- stage V^T tile: rows kb..kb+31, all 128 dims ----
    const __bf16* vrow = Vb + (size_t)(kb + lane) * 1024;
    __builtin_prefetch(vrow + (size_t)32 * 1024, 0, 1);
#pragma unroll
    for (int d0 = 0; d0 < 128; d0 += 8) {
      Pack8 p; p.q = *(const u32x4*)(vrow + d0);
#pragma unroll
      for (int j = 0; j < 8; j++) VtS[wave][d0 + j][lane] = p.e[j];
    }

    // ---- S = Q K^T (two 16x16 n-tiles), K^T B-frags straight from global ----
    v8f s[2] = {};
#pragma unroll
    for (int nt = 0; nt < 2; nt++) {
#pragma unroll
      for (int kc = 0; kc < 4; kc++) {
        Frag16 bk;
        const __bf16* kp = Kb + (size_t)(kb + nt * 16 + l16) * 1024 + kc * 32 + aoff;
        bk.q[0] = *(const u32x4*)kp;
        bk.q[1] = *(const u32x4*)(kp + 16);
        s[nt] = WMMA_BF16(aq[kc].v, bk.v, s[nt]);
      }
    }

    // ---- scale, softcap, causal mask -> LDS ----
#pragma unroll
    for (int nt = 0; nt < 2; nt++) {
#pragma unroll
      for (int r = 0; r < 8; r++) {
        int row   = r + 8 * hi;
        int q_abs = s0 + row;
        int k_abs = kb + nt * 16 + l16;
        float v = s[nt][r] * scale;
        v = 50.0f * tanhf(v * 0.02f);
        v = (k_abs <= q_abs) ? v : -1.0e30f;
        Sbuf[wave][row][nt * 16 + l16] = v;
      }
    }
    __syncthreads();

    // ---- online softmax: lane handles row l16, cols hi*16..hi*16+15 ----
    float sv[16];
    float mloc = -1.0e30f;
#pragma unroll
    for (int j = 0; j < 16; j++) {
      sv[j] = Sbuf[wave][l16][hi * 16 + j];
      mloc = fmaxf(mloc, sv[j]);
    }
    mloc = fmaxf(mloc, __shfl_xor(mloc, 16, 32));
    float m_new = fmaxf(m_r, mloc);
    float alpha = __expf(m_r - m_new);
    float psum = 0.0f;
#pragma unroll
    for (int j = 0; j < 16; j++) {
      float p = __expf(sv[j] - m_new);
      psum += p;
      Pbuf[wave][l16][hi * 16 + j] = (__bf16)p;
    }
    psum += __shfl_xor(psum, 16, 32);
    l_r = l_r * alpha + psum;
    m_r = m_new;
    if (hi == 0) Stats[wave][0][l16] = alpha;
    __syncthreads();

    // ---- rescale O and accumulate P*V ----
    float ar[8];
#pragma unroll
    for (int r = 0; r < 8; r++) ar[r] = Stats[wave][0][r + 8 * hi];
    Frag16 pf;
    pf.q[0] = *(const u32x4*)&Pbuf[wave][l16][aoff];
    pf.q[1] = *(const u32x4*)&Pbuf[wave][l16][16 + aoff];
#pragma unroll
    for (int dt = 0; dt < 8; dt++) {
#pragma unroll
      for (int r = 0; r < 8; r++) o[dt][r] *= ar[r];
      Frag16 bv;
      const __bf16* vp = &VtS[wave][dt * 16 + l16][aoff];
      bv.q[0] = *(const u32x4*)vp;
      bv.q[1] = *(const u32x4*)(vp + 16);
      o[dt] = WMMA_BF16(pf.v, bv.v, o[dt]);
    }
  }

  // ---- finalize: O /= l, write bf16 [B*T][16*128] ----
  if (hi == 0) Stats[wave][1][l16] = 1.0f / l_r;
  __syncthreads();
  float ir[8];
#pragma unroll
  for (int r = 0; r < 8; r++) ir[r] = Stats[wave][1][r + 8 * hi];
#pragma unroll
  for (int dt = 0; dt < 8; dt++) {
#pragma unroll
    for (int r = 0; r < 8; r++) {
      int row = s0 + r + 8 * hi;
      O[(size_t)(b * T + row) * 2048 + qh * 128 + dt * 16 + l16] = (__bf16)(o[dt][r] * ir[r]);
    }
  }
}

// ------------------------------ launch -------------------------------------
extern "C" void kernel_launch(void* const* d_in, const int* in_sizes, int n_in,
                              void* d_out, int out_size, void* d_ws, size_t ws_size,
                              hipStream_t stream) {
  (void)in_sizes; (void)n_in; (void)out_size; (void)ws_size;
  const float* x  = (const float*)d_in[0];
  // d_in[1] = causal mask, regenerated on the fly in the flash kernel
  const float* qk = (const float*)d_in[2];
  const float* kk = (const float*)d_in[3];
  const float* vk = (const float*)d_in[4];
  const float* ok = (const float*)d_in[5];
  float* out = (float*)d_out;

  const int Bq = 2, T = 2048, C = 2048, HQ = 16, KVH = 8, D = 128;
  const int M = Bq * T;  // 4096

  char* ws = (char*)d_ws;
  size_t off = 0;
  auto carve = [&](size_t elems) {
    __bf16* p = (__bf16*)(ws + off);
    off += (elems * sizeof(__bf16) + 255) & ~(size_t)255;
    return p;
  };
  __bf16* xb = carve((size_t)M * C);
  __bf16* wq = carve((size_t)C * (HQ * D));
  __bf16* wk = carve((size_t)C * (KVH * D));
  __bf16* wv = carve((size_t)C * (KVH * D));
  __bf16* wo = carve((size_t)(HQ * D) * C);
  __bf16* Qb = carve((size_t)M * (HQ * D));
  __bf16* Kb = carve((size_t)M * (KVH * D));
  __bf16* Vb = carve((size_t)M * (KVH * D));
  __bf16* Ob = carve((size_t)M * (HQ * D));

  auto cvt = [&](const float* s, __bf16* d, size_t n) {
    int n4 = (int)(n / 4);
    cvt_f32_to_bf16<<<(n4 + 255) / 256, 256, 0, stream>>>(s, d, n4);
  };
  cvt(x,  xb, (size_t)M * C);
  cvt(qk, wq, (size_t)C * HQ * D);
  cvt(kk, wk, (size_t)C * KVH * D);
  cvt(vk, wv, (size_t)C * KVH * D);
  cvt(ok, wo, (size_t)(HQ * D) * C);

  // projections: Q [4096,2048], K/V [4096,1024]
  gemm_bf16_wmma<true><<<dim3(M / 128, (HQ * D) / 128), 256, 0, stream>>>(xb, wq, Qb, HQ * D, C);
  gemm_bf16_wmma<true><<<dim3(M / 128, (KVH * D) / 128), 256, 0, stream>>>(xb, wk, Kb, KVH * D, C);
  gemm_bf16_wmma<true><<<dim3(M / 128, (KVH * D) / 128), 256, 0, stream>>>(xb, wv, Vb, KVH * D, C);

  // RoPE in place
  {
    int nq = M * HQ * 64;
    rope_inplace<<<(nq + 255) / 256, 256, 0, stream>>>(Qb, HQ, M, T);
    int nk = M * KVH * 64;
    rope_inplace<<<(nk + 255) / 256, 256, 0, stream>>>(Kb, KVH, M, T);
  }

  // flash attention: 32 query groups x 32 (b,head) instances
  flash_attn_wmma<<<dim3(T / 64, Bq * HQ), 128, 0, stream>>>(Qb, Kb, Vb, Ob);

  // output projection -> fp32 d_out [4096,2048]
  gemm_bf16_wmma<false><<<dim3(M / 128, C / 128), 256, 0, stream>>>(Ob, wo, out, C, HQ * D);
}
